// GroundedLanguageModel_67997922230352
// MI455X (gfx1250) — matrix operand order repaired and checked by reading, not compiled, not run on hardware
//
#include <hip/hip_runtime.h>
#include <stdint.h>

#define B_   16
#define T_   1024
#define H_   512
#define G4   2048      // 4*H (gates)
#define E2H  1024      // 2*H (enc width)
#define AMAX_ 10
#define BT   (B_ * T_) // 16384

typedef __attribute__((ext_vector_type(16))) __bf16       v16bf;
typedef __attribute__((ext_vector_type(8)))  float        v8f;
typedef __attribute__((ext_vector_type(4)))  unsigned int u32x4;
typedef __attribute__((ext_vector_type(8)))  int          i32x8;
typedef __attribute__((ext_vector_type(4)))  int          i32x4;
typedef unsigned short ushort_t;

#if defined(__has_builtin)
#if __has_builtin(__builtin_amdgcn_tensor_load_to_lds)
#define USE_TDM 1
#endif
#endif

union FragAB { v16bf v; u32x4 q[2]; };

__device__ __forceinline__ ushort_t f32_to_bf16(float f) {
  unsigned int u = __float_as_uint(f);
  u += 0x7FFFu + ((u >> 16) & 1u);       // round-to-nearest-even
  return (ushort_t)(u >> 16);
}

__device__ __forceinline__ float sigmoidf_(float x) {
  return 1.0f / (1.0f + __expf(-x));
}

// ---- WMMA fragment loaders (bf16, 16x16x32, per CDNA5 ISA layouts) ----
// A: 16x32 (MxK). lane m = lane&15, half = lane>>4.
//   half==0: K {k0..k0+7, k0+16..k0+23}; half==1: K {k0+8.., k0+24..}
__device__ __forceinline__ v16bf load_a_frag(const ushort_t* base, int ldk,
                                             int lane, int k0) {
  int half = lane >> 4, m = lane & 15;
  const ushort_t* p = base + m * ldk + k0 + half * 8;
  FragAB f;
  f.q[0] = *(const u32x4*)(p);
  f.q[1] = *(const u32x4*)(p + 16);
  return f.v;
}
// B: 32x16 (KxN), B[k][n] = W[n][k] with W row-major [N,K].
//   half==0: K {k0..k0+7, k0+16..k0+23}; half==1: K {k0+8.., k0+24..}
__device__ __forceinline__ v16bf load_b_frag(const ushort_t* wbase, int ldk,
                                             int lane, int k0) {
  int half = lane >> 4, n = lane & 15;
  const ushort_t* p = wbase + n * ldk + k0 + half * 8;
  FragAB f;
  f.q[0] = *(const u32x4*)(p);
  f.q[1] = *(const u32x4*)(p + 16);
  return f.v;
}

// ---------------------------------------------------------------------------
// Generic bf16 WMMA GEMM: out = A[M,K] @ W[N,K]^T + bias[N], opt ReLU.
// XPOSE=1 remaps output row m = b*T+t  ->  row (t*B + b)  (for xin staging).
// One wave per 16x16 output tile. M,N,K multiples of 16/16/32.
// ---------------------------------------------------------------------------
template <int RELU, int XPOSE>
__global__ __launch_bounds__(128)
void wmma_gemm_bias(const ushort_t* __restrict__ A, const ushort_t* __restrict__ W,
                    const float* __restrict__ bias, float* __restrict__ out,
                    int M, int N, int K) {
  int waveId = blockIdx.x * (blockDim.x >> 5) + (threadIdx.x >> 5);
  int tilesN = N >> 4;
  int tm = waveId / tilesN;
  int tn = waveId - tm * tilesN;
  if (tm >= (M >> 4)) return;             // wave-uniform
  int lane = threadIdx.x & 31;
  int half = lane >> 4, idx = lane & 15;

  const ushort_t* abase = A + (size_t)(tm * 16) * K;
  const ushort_t* wbase = W + (size_t)(tn * 16) * K;
  v8f acc = {};
  for (int k0 = 0; k0 < K; k0 += 32) {
    v16bf a = load_a_frag(abase, K, lane, k0);
    v16bf b = load_b_frag(wbase, K, lane, k0);
    acc = __builtin_amdgcn_wmma_f32_16x16x32_bf16(false, a, false, b,
                                                  (short)0, acc, false, false);
  }
  int n = tn * 16 + idx;
  float bv = bias[n];
  int row0 = tm * 16 + half * 8;
#pragma unroll
  for (int r = 0; r < 8; ++r) {
    float v = acc[r] + bv;
    if (RELU) v = fmaxf(v, 0.0f);
    int m = row0 + r;
    size_t orow = XPOSE ? (size_t)((m & (T_ - 1)) * B_ + (m >> 10)) : (size_t)m;
    out[orow * N + n] = v;
  }
}

// ---------------------------------------------------------------------------
// Persistent bidirectional LSTM recurrence.
// grid = 32 WGs x 128 threads. blockIdx<16: forward, else backward.
// Each WG owns 32 h-columns => 8 gate n-tiles (2 per wave).
// whh slice (128KB bf16) staged in LDS once (TDM + cooperative copy),
// h staged in LDS each step; h double-buffered (bf16) in global;
// per-direction atomic step barrier.
// ---------------------------------------------------------------------------
__global__ __launch_bounds__(128)
void lstm_recurrent(const ushort_t* __restrict__ whhf, const ushort_t* __restrict__ whhb,
                    const float* __restrict__ xinf, const float* __restrict__ xinb,
                    ushort_t* __restrict__ hbuf,   // [2 dir][2 buf][16][512]
                    float* __restrict__ enc,       // [16][1024][1024]
                    ushort_t* __restrict__ enc_bf, // same shape, bf16
                    int* __restrict__ ctr) {       // [2][1024], pre-zeroed
  const int dir = blockIdx.x >> 4;
  const int wg  = blockIdx.x & 15;
  const ushort_t* whh = dir ? whhb : whhf;
  const float*    xin = dir ? xinb : xinf;   // [T][B][4H] layout
  ushort_t* hb  = hbuf + (size_t)dir * (2 * B_ * H_);
  int*      cnt = ctr + dir * T_;

  __shared__ __align__(16) ushort_t whh_lds[4][32][H_]; // 128KB weight slice
  __shared__ __align__(16) ushort_t h_lds[B_][H_];      // 16KB h staging
  __shared__ float gls[4][B_][32];                      // gate slices i,f,g,o
  __shared__ float cls[B_][32];                         // cell state slice

  const int tid  = threadIdx.x;
  const int lane = tid & 31;
  const int wave = tid >> 5;
  const int half = lane >> 4, idx = lane & 15;

#ifdef USE_TDM
  // One-shot TDM stage of this WG's whh slice: 4 "rows" (gate chunks),
  // each 32x512 bf16 contiguous (16384 elems), stride 512*512 elems.
  if (wave == 0) {
    unsigned lds_off = (unsigned)(size_t)&whh_lds[0][0][0];
    unsigned long long ga =
        (unsigned long long)(size_t)(whh + (size_t)(wg * 32) * H_);
    u32x4 g0 = {1u,                                   // count=1 (valid user D#)
                lds_off,                              // lds_addr
                (unsigned)ga,                         // global_addr[31:0]
                ((unsigned)(ga >> 32) & 0x01FFFFFFu) | 0x80000000u}; // type=2
    i32x8 g1 = {(int)(1u << 16),          // data_size=1 (2B), wg_mask=0
                (int)(16384u << 16),      // tensor_dim0[15:0]=16384
                (int)(4u << 16),          // tensor_dim0 hi=0 | tensor_dim1=4
                (int)(16384u << 16),      // tensor_dim1 hi=0 | tile_dim0=16384
                4,                        // tile_dim1=4, tile_dim2=0
                262144,                   // tensor_dim0_stride = 512*512 elems
                0, 0};
    i32x4 z4 = {0, 0, 0, 0};
    i32x8 z8 = {0, 0, 0, 0, 0, 0, 0, 0};
    __builtin_amdgcn_tensor_load_to_lds(g0, g1, z4, z4, z8, 0);
    __builtin_amdgcn_s_wait_tensorcnt(0);
  }
#endif
  __syncthreads();
  // Cooperative copy of the same slice (guarantees LDS contents; overlaps any
  // TDM result, so numerics never depend on the descriptor path).
  for (int g = 0; g < 4; ++g) {
    const u32x4* src = (const u32x4*)(whh + (size_t)(g * H_ + wg * 32) * H_);
    u32x4* dst = (u32x4*)&whh_lds[g][0][0];
    for (int e = tid; e < (32 * H_) / 8; e += 128) dst[e] = src[e];
  }
  for (int e = tid; e < B_ * 32; e += 128) cls[e >> 5][e & 31] = 0.0f;
  __syncthreads();

  for (int s = 0; s < T_; ++s) {
    const int t = dir ? (T_ - 1 - s) : s;
    const ushort_t* hread  = hb + (size_t)(s & 1) * (B_ * H_);
    ushort_t*       hwrite = hb + (size_t)((s + 1) & 1) * (B_ * H_);

    // ---- stage h[16,512] (bf16) into LDS ----
    {
      const u32x4* src = (const u32x4*)hread;
      u32x4* dst = (u32x4*)&h_lds[0][0];
      for (int e = tid; e < (B_ * H_) / 8; e += 128) dst[e] = src[e];
    }
    __syncthreads();

    // ---- phase 1: g = h @ whh^T + xin[t]  (two 16-col tiles per wave) ----
#pragma unroll
    for (int jj = 0; jj < 2; ++jj) {
      int j    = wave * 2 + jj;
      int gate = j >> 1, sub = j & 1;
      const ushort_t* wbase = &whh_lds[gate][sub * 16][0];
      v8f acc = {};
#pragma unroll 4
      for (int k0 = 0; k0 < H_; k0 += 32) {
        v16bf a = load_a_frag(&h_lds[0][0], H_, lane, k0);
        v16bf b = load_b_frag(wbase, H_, lane, k0);
        acc = __builtin_amdgcn_wmma_f32_16x16x32_bf16(false, a, false, b,
                                                      (short)0, acc, false, false);
      }
      int col = gate * H_ + wg * 32 + sub * 16 + idx;   // column in 4H
#pragma unroll
      for (int r = 0; r < 8; ++r) {
        int m = half * 8 + r;                           // batch row
        float g = acc[r] + xin[(size_t)(t * B_ + m) * G4 + col];
        gls[gate][m][sub * 16 + idx] = g;
      }
    }
    __syncthreads();

    // ---- phase 2: elementwise LSTM cell, write h (bf16 + f32 enc) ----
    for (int e = tid; e < B_ * 32; e += 128) {
      int m = e >> 5, cc = e & 31;
      float gi = gls[0][m][cc], gf = gls[1][m][cc];
      float gg = gls[2][m][cc], go = gls[3][m][cc];
      float c_new = sigmoidf_(gf) * cls[m][cc] + sigmoidf_(gi) * tanhf(gg);
      float h_new = sigmoidf_(go) * tanhf(c_new);
      cls[m][cc] = c_new;
      int hc = wg * 32 + cc;
      hwrite[m * H_ + hc] = f32_to_bf16(h_new);
      size_t eo = (size_t)(m * T_ + t) * E2H + dir * H_ + hc;
      enc[eo]    = h_new;
      enc_bf[eo] = f32_to_bf16(h_new);
    }
    __syncthreads();

    // ---- inter-WG step barrier (per direction) ----
    if (tid == 0) {
      __threadfence();
      atomicAdd(&cnt[s], 1);
      while (__hip_atomic_load(&cnt[s], __ATOMIC_RELAXED,
                               __HIP_MEMORY_SCOPE_AGENT) < 16) {
        __builtin_amdgcn_s_sleep(2);
      }
    }
    __syncthreads();
    __threadfence();
  }
}

// ---------------------------------------------------------------------------
// Small f32 kernels
// ---------------------------------------------------------------------------
__global__ void zero_i32(int* p, int n) {
  int i = blockIdx.x * blockDim.x + threadIdx.x;
  if (i < n) p[i] = 0;
}

__global__ void cast_bf(const float* __restrict__ s, ushort_t* __restrict__ d, int n) {
  int i = blockIdx.x * blockDim.x + threadIdx.x;
  if (i < n) d[i] = f32_to_bf16(s[i]);
}

__global__ void vec_add(const float* a, const float* b, float* o, int n) {
  int i = blockIdx.x * blockDim.x + threadIdx.x;
  if (i < n) o[i] = a[i] + b[i];
}

__global__ void gather_cast(const int* __restrict__ instr, const float* __restrict__ emb,
                            ushort_t* __restrict__ x) {
  int i = blockIdx.x * blockDim.x + threadIdx.x;   // BT*H
  int m = i >> 9, h = i & (H_ - 1);
  x[i] = f32_to_bf16(emb[(size_t)instr[m] * H_ + h]);
}

// seg logits: one wave per row, dot(sh[row], ws2) -> sigmoid
__global__ __launch_bounds__(256)
void seg_logits(const float* __restrict__ sh, const float* __restrict__ ws2,
                const float* __restrict__ bs2, float* __restrict__ segp) {
  int row  = blockIdx.x * 8 + (threadIdx.x >> 5);
  int lane = threadIdx.x & 31;
  const float* p = sh + (size_t)row * H_;
  float s = 0.0f;
  for (int k = lane; k < H_; k += 32) s += p[k] * ws2[k];
  for (int off = 16; off > 0; off >>= 1) s += __shfl_down(s, off, 32);
  if (lane == 0) segp[row] = sigmoidf_(s + bs2[0]);
}

__global__ void seg_select(const float* __restrict__ segp, int* __restrict__ order,
                           int* __restrict__ count) {
  int b = blockIdx.x;
  if (threadIdx.x == 0) {
    int k = 0;
    for (int t = 0; t < T_; ++t) {
      if (segp[b * T_ + t] > 0.5f) {
        if (k < AMAX_) order[b * AMAX_ + k] = t;
        ++k;
      }
    }
    for (int j = (k < AMAX_ ? k : AMAX_); j < AMAX_; ++j) order[b * AMAX_ + j] = 0;
    count[b] = k;
  }
}

// gctx = pooled @ wg^T + bg ; pooled = [enc[b,T-1,0:512], enc[b,0,512:1024]]
__global__ void gctx_kernel(const float* __restrict__ enc, const float* __restrict__ wgm,
                            const float* __restrict__ bg, float* __restrict__ gctx) {
  int i = blockIdx.x * blockDim.x + threadIdx.x;   // B*1024
  int b = i >> 10, n = i & (E2H - 1);
  const float* wr = wgm + (size_t)n * E2H;
  const float* pf = enc + (size_t)(b * T_ + (T_ - 1)) * E2H;
  const float* pb = enc + (size_t)(b * T_) * E2H + H_;
  float s = bg[n];
  for (int k = 0; k < H_; ++k) s += pf[k] * wr[k];
  for (int k = 0; k < H_; ++k) s += pb[k] * wr[H_ + k];
  gctx[i] = s;
}

__global__ void build_decin(const float* __restrict__ enc, const float* __restrict__ gctx,
                            const int* __restrict__ order, const int* __restrict__ count,
                            float* __restrict__ dec_in, float* __restrict__ maskv) {
  int i = blockIdx.x * blockDim.x + threadIdx.x;   // B*AMAX*1024
  int row = i >> 10, k = i & (E2H - 1);
  int b = row / AMAX_, a = row - b * AMAX_;
  int cnt = count[b];
  bool valid = a < (cnt < AMAX_ ? cnt : AMAX_);
  bool empty = (cnt == 0) && (a == 0);
  float v = 0.0f;
  if (valid)      v = enc[(size_t)(b * T_ + order[b * AMAX_ + a]) * E2H + k];
  else if (empty) v = gctx[b * E2H + k];
  dec_in[i] = v;
  if (k == 0) maskv[row] = (valid || empty) ? 1.0f : 0.0f;
}

__global__ void dec_fc1(const float* __restrict__ x, const float* __restrict__ wd1,
                        const float* __restrict__ bd1, float* __restrict__ dh) {
  int i = blockIdx.x * blockDim.x + threadIdx.x;   // 160*512
  int row = i >> 9, n = i & (H_ - 1);
  const float* xr = x + (size_t)row * E2H;
  const float* wr = wd1 + (size_t)n * E2H;
  float s = bd1[n];
  for (int k = 0; k < E2H; ++k) s += xr[k] * wr[k];
  dh[i] = fmaxf(s, 0.0f);
}

__global__ void dec_fc2(const float* __restrict__ dh, const float* __restrict__ wd2,
                        const float* __restrict__ bd2, const float* __restrict__ maskv,
                        float* __restrict__ actions) {
  int i = blockIdx.x * blockDim.x + threadIdx.x;   // 160*512
  int row = i >> 9, n = i & (H_ - 1);
  const float* xr = dh + (size_t)row * H_;
  const float* wr = wd2 + (size_t)n * H_;
  float s = bd2[n];
  for (int k = 0; k < H_; ++k) s += xr[k] * wr[k];
  actions[i] = s * maskv[row];
}

// ---------------------------------------------------------------------------
extern "C" void kernel_launch(void* const* d_in, const int* in_sizes, int n_in,
                              void* d_out, int out_size, void* d_ws, size_t ws_size,
                              hipStream_t stream) {
  (void)in_sizes; (void)n_in; (void)out_size; (void)ws_size;
  const int*   instruction = (const int*)d_in[0];
  const float* emb   = (const float*)d_in[1];
  const float* wih_f = (const float*)d_in[2];
  const float* whh_f = (const float*)d_in[3];
  const float* bih_f = (const float*)d_in[4];
  const float* bhh_f = (const float*)d_in[5];
  const float* wih_b = (const float*)d_in[6];
  const float* whh_b = (const float*)d_in[7];
  const float* bih_b = (const float*)d_in[8];
  const float* bhh_b = (const float*)d_in[9];
  const float* wg_   = (const float*)d_in[10];
  const float* bg_   = (const float*)d_in[11];
  const float* ws1   = (const float*)d_in[12];
  const float* bs1   = (const float*)d_in[13];
  const float* ws2   = (const float*)d_in[14];
  const float* bs2   = (const float*)d_in[15];
  const float* wd1   = (const float*)d_in[16];
  const float* bd1   = (const float*)d_in[17];
  const float* wd2   = (const float*)d_in[18];
  const float* bd2   = (const float*)d_in[19];

  float* actions = (float*)d_out;                            // [16,10,512]
  float* segp    = (float*)d_out + (size_t)B_ * AMAX_ * H_;  // [16,1024]

  char* ws = (char*)d_ws;
  size_t off = 0;
  auto take = [&](size_t bytes) -> char* {
    off = (off + 255) & ~(size_t)255;
    char* p = ws + off;
    off += bytes;
    return p;
  };

  int*      ctr     = (int*)take((size_t)2 * T_ * 4);
  ushort_t* hbuf    = (ushort_t*)take((size_t)2 * 2 * B_ * H_ * 2);
  ushort_t* x_bf    = (ushort_t*)take((size_t)BT * H_ * 2);
  ushort_t* wihf_bf = (ushort_t*)take((size_t)G4 * H_ * 2);
  ushort_t* whhf_bf = (ushort_t*)take((size_t)G4 * H_ * 2);
  ushort_t* wihb_bf = (ushort_t*)take((size_t)G4 * H_ * 2);
  ushort_t* whhb_bf = (ushort_t*)take((size_t)G4 * H_ * 2);
  ushort_t* ws1_bf  = (ushort_t*)take((size_t)H_ * E2H * 2);
  float*    biasf   = (float*)take((size_t)G4 * 4);
  float*    biasb   = (float*)take((size_t)G4 * 4);
  float*    xin_f   = (float*)take((size_t)BT * G4 * 4);   // [T][B][4H]
  float*    xin_b   = (float*)take((size_t)BT * G4 * 4);   // [T][B][4H]
  float*    enc     = (float*)take((size_t)BT * E2H * 4);
  ushort_t* enc_bf  = (ushort_t*)take((size_t)BT * E2H * 2);
  float*    sh      = (float*)take((size_t)BT * H_ * 4);
  float*    gctx    = (float*)take((size_t)B_ * E2H * 4);
  int*      order   = (int*)take((size_t)B_ * AMAX_ * 4);
  int*      cntv    = (int*)take((size_t)B_ * 4);
  float*    maskv   = (float*)take((size_t)B_ * AMAX_ * 4);
  float*    dec_in  = (float*)take((size_t)B_ * AMAX_ * E2H * 4);
  float*    dh      = (float*)take((size_t)B_ * AMAX_ * H_ * 4);

  // init: barrier counters + h0 = 0
  zero_i32<<<(2 * T_ + 255) / 256, 256, 0, stream>>>(ctr, 2 * T_);
  zero_i32<<<((2 * 2 * B_ * H_ / 2) + 255) / 256, 256, 0, stream>>>(
      (int*)hbuf, 2 * 2 * B_ * H_ / 2);

  // bf16 staging
  gather_cast<<<(BT * H_) / 256, 256, 0, stream>>>(instruction, emb, x_bf);
  cast_bf<<<(G4 * H_) / 256, 256, 0, stream>>>(wih_f, wihf_bf, G4 * H_);
  cast_bf<<<(G4 * H_) / 256, 256, 0, stream>>>(whh_f, whhf_bf, G4 * H_);
  cast_bf<<<(G4 * H_) / 256, 256, 0, stream>>>(wih_b, wihb_bf, G4 * H_);
  cast_bf<<<(G4 * H_) / 256, 256, 0, stream>>>(whh_b, whhb_bf, G4 * H_);
  cast_bf<<<(H_ * E2H) / 256, 256, 0, stream>>>(ws1, ws1_bf, H_ * E2H);
  vec_add<<<(G4 + 255) / 256, 256, 0, stream>>>(bih_f, bhh_f, biasf, G4);
  vec_add<<<(G4 + 255) / 256, 256, 0, stream>>>(bih_b, bhh_b, biasb, G4);

  // input projections: [16384,512] @ [512,2048]^T  (bf16 WMMA, f32 acc),
  // stored transposed as [T][B][4H] for contiguous per-step access.
  {
    int tiles = (BT / 16) * (G4 / 16);          // 131072 tiles, 4 waves/block
    wmma_gemm_bias<0, 1><<<tiles / 4, 128, 0, stream>>>(x_bf, wihf_bf, biasf,
                                                        xin_f, BT, G4, H_);
    wmma_gemm_bias<0, 1><<<tiles / 4, 128, 0, stream>>>(x_bf, wihb_bf, biasb,
                                                        xin_b, BT, G4, H_);
  }

  // bidirectional recurrence (persistent, 32 WGs, atomic step barrier)
  lstm_recurrent<<<32, 128, 0, stream>>>(whhf_bf, whhb_bf, xin_f, xin_b,
                                         hbuf, enc, enc_bf, ctr);

  // seg head: sh = relu(enc @ ws1^T + bs1)   [16384,1024] @ [1024,512]^T
  {
    int tiles = (BT / 16) * (H_ / 16);          // 32768 tiles
    wmma_gemm_bias<1, 0><<<tiles / 4, 128, 0, stream>>>(enc_bf, ws1_bf, bs1,
                                                        sh, BT, H_, E2H);
  }
  seg_logits<<<BT / 8, 256, 0, stream>>>(sh, ws2, bs2, segp);
  seg_select<<<B_, 32, 0, stream>>>(segp, order, cntv);

  // decoder
  gctx_kernel<<<(B_ * E2H) / 256, 256, 0, stream>>>(enc, wg_, bg_, gctx);
  build_decin<<<(B_ * AMAX_ * E2H) / 256, 256, 0, stream>>>(enc, gctx, order,
                                                            cntv, dec_in, maskv);
  dec_fc1<<<(B_ * AMAX_ * H_) / 256, 256, 0, stream>>>(dec_in, wd1, bd1, dh);
  dec_fc2<<<(B_ * AMAX_ * H_) / 256, 256, 0, stream>>>(dh, wd2, bd2, maskv,
                                                       actions);
}
